// KoLeoLoss_75642964017420
// MI455X (gfx1250) — matrix operand order repaired and checked by reading, not compiled, not run on hardware
//
#include <hip/hip_runtime.h>
#include <hip/hip_bf16.h>

typedef __attribute__((ext_vector_type(16))) __bf16        v16bf;
typedef __attribute__((ext_vector_type(8)))  __bf16        v8bf;
typedef __attribute__((ext_vector_type(8)))  float         v8f;
typedef __attribute__((ext_vector_type(4)))  unsigned int  u32x4;

#define DIM   256
#define KEPS  1e-8f
#define NCHUNK 8            // column-range split for occupancy
#define WAVES_PER_WG 4      // 4 waves x 16 rows = 64 rows per WG
#define LDS_ROW_Q 33        // 33 u32x4 = 528B per tile row (16B pad -> bank-conflict floor)

// ---------------- Kernel 1: row L2-normalize; write fp32 + bf16 copies ------
__global__ void koleo_normalize(const float* __restrict__ x,
                                float* __restrict__ xn,
                                unsigned short* __restrict__ xbf_u16) {
    __shared__ float s[DIM];
    const int row = blockIdx.x;
    const int d   = threadIdx.x;
    float v = x[(size_t)row * DIM + d];
    s[d] = v * v;
    __syncthreads();
    for (int off = DIM / 2; off > 0; off >>= 1) {
        if (d < off) s[d] += s[d + off];
        __syncthreads();
    }
    const float inv = 1.0f / fmaxf(sqrtf(s[0]), KEPS);
    const float xv  = v * inv;
    xn[(size_t)row * DIM + d] = xv;
    ((__bf16*)xbf_u16)[(size_t)row * DIM + d] = (__bf16)xv;
}

// ---------------- Kernel 2: WMMA GEMM + fused argmax, LDS-staged B ----------
// grid = (n/64 row groups, NCHUNK column chunks), block = 128 threads.
// Each wave: one 16-row strip, A (16x256 bf16) resident in 64 VGPRs.
// B tiles staged cooperatively into double-buffered LDS (8KB/tile); all 8
// B operands of a tile are preloaded into registers (one DS clause, one wait)
// so the 8 chained WMMAs issue back-to-back. One barrier per tile suffices:
// reads of a buffer are dscnt-waited (consumed by WMMA) before the wave
// reaches the next barrier, and the conflicting rewrite is 2 iterations away.
__global__ void __launch_bounds__(128)
koleo_argmax(const unsigned short* __restrict__ xbf_u16,
             float* __restrict__ pval, int* __restrict__ pidx, int n) {
    const __bf16* xbf = (const __bf16*)xbf_u16;
    const int tid  = threadIdx.x;
    const int lane = tid & 31;
    const int wave = tid >> 5;
    const int half = lane >> 4;       // 0: lanes 0-15, 1: lanes 16-31
    const int lm   = lane & 15;
    const int m0   = blockIdx.x * (16 * WAVES_PER_WG) + wave * 16;
    const int chunk = blockIdx.y;
    const int tilesPerChunk = (n >> 4) / NCHUNK;
    const int jt0 = chunk * tilesPerChunk;

    __shared__ u32x4 lsB[2][16 * LDS_ROW_Q];   // 2 x 8.25KB double buffer

    // A operand, 16-bit 16x32 layout per K-step of 32:
    //   lanes 0-15 : VGPR0-3 = K 0..7,  VGPR4-7 = K 16..23   (row M = lane)
    //   lanes 16-31: VGPR0-3 = K 8..15, VGPR4-7 = K 24..31
    union AV { v16bf v; v8bf h[2]; };
    v16bf a[8];
    {
        const __bf16* rowp = xbf + (size_t)(m0 + lm) * DIM;
        #pragma unroll
        for (int k = 0; k < 8; ++k) {
            AV t;
            t.h[0] = *(const v8bf*)(rowp + k * 32 + half * 8);
            t.h[1] = *(const v8bf*)(rowp + k * 32 + 16 + half * 8);
            a[k] = t.v;
        }
    }

    float bestv[8];
    int   besti[8];
    #pragma unroll
    for (int r = 0; r < 8; ++r) { bestv[r] = -2.0f; besti[r] = 0; }

    // Staging role: thread t loads 64B of the 16x512B tile (row sr, piece sc).
    const int sr = tid >> 3;     // 0..15
    const int sc = tid & 7;      // 0..7 (64B pieces)

    u32x4 st0, st1, st2, st3;    // in-flight tile (prefetch regs)
    {
        const u32x4* src = (const u32x4*)(xbf + (size_t)(jt0 * 16 + sr) * DIM + sc * 32);
        st0 = src[0]; st1 = src[1]; st2 = src[2]; st3 = src[3];
    }

    union BV { v16bf v; u32x4 q[2]; };

    for (int t = 0; t < tilesPerChunk; ++t) {
        const int buf = t & 1;
        // commit prefetched tile to LDS (row stride 528B: conflict-optimal)
        {
            u32x4* dst = &lsB[buf][sr * LDS_ROW_Q + sc * 4];
            dst[0] = st0; dst[1] = st1; dst[2] = st2; dst[3] = st3;
        }
        __syncthreads();
        // issue next tile's global loads now; they overlap the WMMA chain
        if (t + 1 < tilesPerChunk) {
            const u32x4* src =
                (const u32x4*)(xbf + (size_t)((jt0 + t + 1) * 16 + sr) * DIM + sc * 32);
            st0 = src[0]; st1 = src[1]; st2 = src[2]; st3 = src[3];
        }
        // Preload all 8 B operands (16 ds_load_b128 in one clause, one wait).
        // B operand, 32x16: lane = column N, same K layout as A (half h: K 8h..8h+7, 16+8h..16+8h+7).
        const int bbase = lm * LDS_ROW_Q + half;
        BV b[8];
        #pragma unroll
        for (int k = 0; k < 8; ++k) {
            b[k].q[0] = lsB[buf][bbase + k * 4];
            b[k].q[1] = lsB[buf][bbase + k * 4 + 2];
        }
        v8f c = {0.f, 0.f, 0.f, 0.f, 0.f, 0.f, 0.f, 0.f};
        #pragma unroll
        for (int k = 0; k < 8; ++k) {
            c = __builtin_amdgcn_wmma_f32_16x16x32_bf16(
                    false, a[k], false, b[k].v, (short)0, c, false, false);
        }
        // C/D layout: VGPR r, lanes 0-15 -> (M=r, N=lane); lanes 16-31 -> M=8+r
        const int col = (jt0 + t) * 16 + lm;
        #pragma unroll
        for (int r = 0; r < 8; ++r) {
            const int   row  = m0 + r + half * 8;
            const float v    = c[r];
            const bool  take = (col != row) & (v > bestv[r]);
            bestv[r] = take ? v   : bestv[r];
            besti[r] = take ? col : besti[r];
        }
        // no trailing barrier needed: double buffer + dscnt-consumed reads
    }

    // Reduce (max, idx) across the 16 lanes of each half-wave; write partials.
    #pragma unroll
    for (int r = 0; r < 8; ++r) {
        float v  = bestv[r];
        int   id = besti[r];
        #pragma unroll
        for (int off = 8; off >= 1; off >>= 1) {
            float v2 = __shfl_xor(v,  off, 32);
            int   i2 = __shfl_xor(id, off, 32);
            const bool take = (v2 > v) | ((v2 == v) & (i2 < id));
            v  = take ? v2 : v;
            id = take ? i2 : id;
        }
        if (lm == 0) {
            const int row = m0 + r + half * 8;
            pval[(size_t)row * NCHUNK + chunk] = v;
            pidx[(size_t)row * NCHUNK + chunk] = id;
        }
    }
}

// ---------------- Kernel 2b: merge per-chunk argmax partials ----------------
__global__ void koleo_merge(const float* __restrict__ pval,
                            const int* __restrict__ pidx,
                            int* __restrict__ argi, int n) {
    const int i = blockIdx.x * 256 + threadIdx.x;
    if (i >= n) return;
    float v  = -1e30f;
    int   id = 0;
    // chunks cover ascending column ranges -> strict > keeps smallest index
    for (int c = 0; c < NCHUNK; ++c) {
        const float v2 = pval[(size_t)i * NCHUNK + c];
        const int   i2 = pidx[(size_t)i * NCHUNK + c];
        const bool take = (v2 > v) | ((v2 == v) & (i2 < id));
        v  = take ? v2 : v;
        id = take ? i2 : id;
    }
    argi[i] = id;
}

// ---------------- Kernel 3: exact fp32 per-row log-distance -----------------
__global__ void koleo_rowdist(const float* __restrict__ xn,
                              const int* __restrict__ argi,
                              float* __restrict__ logd) {
    __shared__ float s[DIM];
    const int i = blockIdx.x;
    const int d = threadIdx.x;
    const int I = argi[i];
    const float diff = xn[(size_t)i * DIM + d] - xn[(size_t)I * DIM + d] + KEPS;
    s[d] = diff * diff;
    __syncthreads();
    for (int off = DIM / 2; off > 0; off >>= 1) {
        if (d < off) s[d] += s[d + off];
        __syncthreads();
    }
    if (d == 0) logd[i] = logf(sqrtf(s[0]) + KEPS);
}

// ---------------- Kernel 4: deterministic mean ------------------------------
__global__ void koleo_mean(const float* __restrict__ logd,
                           float* __restrict__ out, int n) {
    __shared__ float s[256];
    const int t = threadIdx.x;
    float acc = 0.f;
    for (int j = t; j < n; j += 256) acc += logd[j];   // fixed order per thread
    s[t] = acc;
    __syncthreads();
    for (int off = 128; off > 0; off >>= 1) {
        if (t < off) s[t] += s[t + off];
        __syncthreads();
    }
    if (t == 0) out[0] = -s[0] / (float)n;
}

extern "C" void kernel_launch(void* const* d_in, const int* in_sizes, int n_in,
                              void* d_out, int out_size, void* d_ws, size_t ws_size,
                              hipStream_t stream) {
    const float* student = (const float*)d_in[0];
    const int rows_total = in_sizes[0] / DIM;   // 81920
    const int n = rows_total / 10;              // NCROPS=10 -> 8192

    // Workspace layout:
    //   xn   : n*256 fp32   (8 MB)
    //   xbf  : n*256 bf16   (4 MB)
    //   argi : n int32
    //   logd : n fp32
    //   pval : n*NCHUNK fp32  (256 KB)
    //   pidx : n*NCHUNK int32 (256 KB)
    char* w = (char*)d_ws;
    float*          xn   = (float*)w;
    unsigned short* xbf  = (unsigned short*)(w + (size_t)n * DIM * sizeof(float));
    char*           p    = (char*)xbf + (size_t)n * DIM * sizeof(unsigned short);
    int*            argi = (int*)p;               p += (size_t)n * sizeof(int);
    float*          logd = (float*)p;             p += (size_t)n * sizeof(float);
    float*          pval = (float*)p;             p += (size_t)n * NCHUNK * sizeof(float);
    int*            pidx = (int*)p;
    float*          out  = (float*)d_out;

    koleo_normalize<<<n, DIM, 0, stream>>>(student, xn, xbf);

    dim3 g2(n / (16 * WAVES_PER_WG), NCHUNK);
    koleo_argmax<<<g2, 128, 0, stream>>>(xbf, pval, pidx, n);

    koleo_merge  <<<(n + 255) / 256, 256, 0, stream>>>(pval, pidx, argi, n);
    koleo_rowdist<<<n, DIM, 0, stream>>>(xn, argi, logd);
    koleo_mean   <<<1, 256, 0, stream>>>(logd, out, n);
}
